// GRULayer_72181220377136
// MI455X (gfx1250) — matrix operand order repaired and checked by reading, not compiled, not run on hardware
//
#include <hip/hip_runtime.h>
#include <math.h>

// GRU: B=64, T=512, D=512, H=1024.  bf16 WMMA (16x16x32) persistent-RNN kernel.
constexpr int BB   = 64;      // batch
constexpr int TT   = 512;     // time
constexpr int DD   = 512;     // input dim
constexpr int HH   = 1024;    // hidden dim
constexpr int KTOT = DD + HH; // 1536 (GEMM K)
constexpr int NGATE = 2 * HH; // 2048
constexpr int GRID_WGS = 64;
constexpr int NTHREADS = 256; // 8 wave32 waves

typedef __attribute__((ext_vector_type(16))) __bf16 v16bf;
typedef __attribute__((ext_vector_type(8)))  float  v8f;

// ---------------- workspace layout (bytes) ----------------
constexpr size_t OFF_WGT   = 0;                                   // 2048*1536 bf16 (N-major, K contiguous)
constexpr size_t SZ_WGT    = (size_t)NGATE * KTOT * 2;
constexpr size_t OFF_WCT   = OFF_WGT + SZ_WGT;                    // 1024*1536 bf16
constexpr size_t SZ_WCT    = (size_t)HH * KTOT * 2;
constexpr size_t OFF_XSWZ  = OFF_WCT + SZ_WCT;                    // B*T*D bf16, A-fragment swizzled per t
constexpr size_t SZ_XSWZ   = (size_t)BB * TT * DD * 2;
constexpr size_t OFF_HSWZ  = OFF_XSWZ + SZ_XSWZ;                  // 4 mt * 32 kc * 512 bf16
constexpr size_t SZ_HSWZ   = (size_t)4 * 32 * 512 * 2;
constexpr size_t OFF_RHSWZ = OFF_HSWZ + SZ_HSWZ;
constexpr size_t SZ_RHSWZ  = SZ_HSWZ;
constexpr size_t OFF_HBUF  = OFF_RHSWZ + SZ_RHSWZ;                // B*H f32
constexpr size_t SZ_HBUF   = (size_t)BB * HH * 4;
constexpr size_t OFF_UBUF  = OFF_HBUF + SZ_HBUF;                  // B*H f32 (update gate u)
constexpr size_t SZ_UBUF   = SZ_HBUF;
constexpr size_t OFF_BAR   = OFF_UBUF + SZ_UBUF;                  // barrier counter + generation

__device__ __forceinline__ v8f wmma_bf16(v16bf a, v16bf b, v8f c) {
  // D = A(16x32 bf16) x B(32x16 bf16) + C(16x16 f32)
  return __builtin_amdgcn_wmma_f32_16x16x32_bf16(false, a, false, b, (short)0, c, false, false);
}

__device__ __forceinline__ v16bf ldfrag(const __bf16* p) {
  return *(const v16bf*)p;  // 32B aligned by construction
}

// Scatter one value into this file's 16-bit A-fragment layout (contiguous k order, matching the B loads):
// lane = m + (k&16 ? 16 : 0), elem = k&15  for a 16x32 tile.
__device__ __forceinline__ size_t a_swizzle_idx(int m, int kglob) {
  int kc   = kglob >> 5;          // 32-wide K chunk
  int kl   = kglob & 31;
  int lane = (m & 15) + ((kl & 16) ? 16 : 0);
  int elem = kl & 15;
  return ((size_t)((m >> 4) * 32 + kc)) * 512 + (size_t)lane * 16 + elem;
}

__device__ __forceinline__ void grid_sync(unsigned* cnt, unsigned* gen) {
  __threadfence();
  __syncthreads();
  if (threadIdx.x == 0) {
    unsigned g = __hip_atomic_load(gen, __ATOMIC_RELAXED, __HIP_MEMORY_SCOPE_AGENT);
    unsigned a = __hip_atomic_fetch_add(cnt, 1u, __ATOMIC_ACQ_REL, __HIP_MEMORY_SCOPE_AGENT);
    if (a == (unsigned)(GRID_WGS - 1)) {
      __hip_atomic_store(cnt, 0u, __ATOMIC_RELAXED, __HIP_MEMORY_SCOPE_AGENT);
      __hip_atomic_store(gen, g + 1u, __ATOMIC_RELEASE, __HIP_MEMORY_SCOPE_AGENT);
    } else {
      while (__hip_atomic_load(gen, __ATOMIC_ACQUIRE, __HIP_MEMORY_SCOPE_AGENT) == g) {
        __builtin_amdgcn_s_sleep(2);
      }
    }
  }
  __syncthreads();
}

// ---------------- setup kernels ----------------
__global__ void gru_init(float* hbuf, __bf16* hswz, unsigned* bar) {
  int i = blockIdx.x * blockDim.x + threadIdx.x;
  if (i < BB * HH) { hbuf[i] = 0.0f; hswz[i] = (__bf16)0.0f; }
  if (i < 2) bar[i] = 0u;
}

// Transpose weights to N-major / K-contiguous bf16 so a B-fragment is one
// contiguous 32B load per lane: lane l holds col n=l%16, k = (l>=16?16:0)+i.
__global__ void gru_prep_weights(const float* __restrict__ gk, const float* __restrict__ ck,
                                 __bf16* __restrict__ WgT, __bf16* __restrict__ WcT) {
  int i = blockIdx.x * blockDim.x + threadIdx.x;
  const int tg = NGATE * KTOT;
  if (i < tg) {
    int n = i / KTOT, k = i - n * KTOT;
    WgT[i] = (__bf16)gk[(size_t)k * NGATE + n];
  } else {
    int j = i - tg;
    if (j < HH * KTOT) {
      int n = j / KTOT, k = j - n * KTOT;
      WcT[j] = (__bf16)ck[(size_t)k * HH + n];
    }
  }
}

// Pre-swizzle X[B,T,D] -> bf16 A-fragment layout per timestep.
__global__ void gru_swizzle_x(const float* __restrict__ X, __bf16* __restrict__ xswz) {
  int o = blockIdx.x * blockDim.x + threadIdx.x;  // < B*T*D = 2^24
  int i    = o & 15;
  int lane = (o >> 4) & 31;
  int kc   = (o >> 9) & 15;   // D/32 = 16 chunks
  int mt   = (o >> 13) & 3;   // 4 M tiles (B=64)
  int t    = o >> 15;
  int m = (mt << 4) + (lane & 15);
  int k = (kc << 5) + ((lane >> 4) << 4) + i;
  xswz[o] = (__bf16)X[((size_t)m * TT + t) * DD + k];
}

// ---------------- persistent recurrent kernel ----------------
__global__ void __launch_bounds__(NTHREADS, 1)
gru_persistent(const __bf16* __restrict__ xswz,
               const __bf16* __restrict__ WgT,
               const __bf16* __restrict__ WcT,
               const float*  __restrict__ gbias,
               const float*  __restrict__ cbias,
               float*        __restrict__ hbuf,
               __bf16*       __restrict__ hswz,
               __bf16*       __restrict__ rhswz,
               float*        __restrict__ ubuf,
               unsigned*                  bar,
               float*        __restrict__ out) {
  const int wg   = blockIdx.x;          // 64 workgroups
  const int tid  = threadIdx.x;
  const int lane = tid & 31;            // wave32
  const int wv   = tid >> 5;            // 8 waves
  const int nloc = lane & 15;
  const int hi   = (lane >> 4) << 3;    // C/D row offset: 0 or 8
  const int kh   = (lane >> 4) << 4;    // B-frag K half: 0 or 16

  unsigned* cnt = bar;
  unsigned* gen = bar + 1;

  const int mt  = wv & 3;               // M tile (shared by both phases)
  const int ntg = wv >> 2;              // gates N tile within WG slice (0..1)
  const int n0g = wg * 32;              // gates cols [n0g, n0g+32)
  const int n0c = wg * 16;              // cand  cols [n0c, n0c+16)

  __shared__ float red[4 * 256];        // split-K partial sums for cand phase

  for (int t = 0; t < TT; ++t) {
    // ================= Phase G: gates = sigmoid([x,h] @ Wg + bg) =============
    v8f acc = {};
    const __bf16* xA = xswz + ((size_t)(t * 4 + mt) * 16) * 512 + lane * 16;
    const __bf16* hA = hswz + (size_t)mt * 32 * 512 + lane * 16;
    const __bf16* bg = WgT + (size_t)(n0g + ntg * 16 + nloc) * KTOT + kh;
    __builtin_prefetch(bg + 512, 0, 1);   // pull h-part weight row toward L2/L0
#pragma unroll 4
    for (int kc = 0; kc < 16; ++kc) {     // x part: K = 0..511
      v16bf a = ldfrag(xA + kc * 512);
      v16bf b = ldfrag(bg + kc * 32);
      acc = wmma_bf16(a, b, acc);
    }
#pragma unroll 4
    for (int kc = 0; kc < 32; ++kc) {     // h part: K = 512..1535
      v16bf a = ldfrag(hA + kc * 512);
      v16bf b = ldfrag(bg + 512 + kc * 32);
      acc = wmma_bf16(a, b, acc);
    }
    {
      const int n = n0g + ntg * 16 + nloc;
      const float bias = gbias[n];
#pragma unroll
      for (int j = 0; j < 8; ++j) {
        int m = (mt << 4) + j + hi;
        float g = 1.0f / (1.0f + __expf(-(acc[j] + bias)));
        if (n < HH) {                     // reset gate -> scatter r*h (bf16, A-layout)
          float rh = g * hbuf[m * HH + n];
          rhswz[a_swizzle_idx(m, n)] = (__bf16)rh;
        } else {                          // update gate -> stash u (f32)
          ubuf[m * HH + (n - HH)] = g;
        }
      }
    }
    grid_sync(cnt, gen);

    // ====== Phase C: c = tanh([x, r*h] @ Wc + bc); h = u*h + (1-u)*c =========
    v8f accc = {};
    const __bf16* bc = WcT + (size_t)(n0c + nloc) * KTOT + kh;
    const __bf16* rA = rhswz + (size_t)mt * 32 * 512 + lane * 16;
    __builtin_prefetch(bc + 512, 0, 1);
    const int xk0 = (wv < 4) ? 0 : 8;     // split-K across wave pairs
    const int rk0 = (wv < 4) ? 0 : 16;
#pragma unroll 4
    for (int kc = xk0; kc < xk0 + 8; ++kc) {
      v16bf a = ldfrag(xA + kc * 512);
      v16bf b = ldfrag(bc + kc * 32);
      accc = wmma_bf16(a, b, accc);
    }
#pragma unroll 4
    for (int kc = rk0; kc < rk0 + 16; ++kc) {
      v16bf a = ldfrag(rA + kc * 512);
      v16bf b = ldfrag(bc + 512 + kc * 32);
      accc = wmma_bf16(a, b, accc);
    }
    if (wv >= 4) {                        // high waves publish partials via LDS
#pragma unroll
      for (int j = 0; j < 8; ++j) red[(wv - 4) * 256 + lane * 8 + j] = accc[j];
    }
    __syncthreads();
    if (wv < 4) {
      const int n = n0c + nloc;
      const float bias = cbias[n];
#pragma unroll
      for (int j = 0; j < 8; ++j) {
        float v = accc[j] + red[wv * 256 + lane * 8 + j] + bias;
        float e = __expf(-2.0f * v);
        float c = (1.0f - e) / (1.0f + e);          // tanh
        int m = (mt << 4) + j + hi;
        float u = ubuf[m * HH + n];
        float hn = u * hbuf[m * HH + n] + (1.0f - u) * c;
        hbuf[m * HH + n] = hn;
        out[((size_t)m * TT + t) * HH + n] = hn;
        hswz[a_swizzle_idx(m, n)] = (__bf16)hn;     // bf16 A-layout for next step
      }
    }
    grid_sync(cnt, gen);
  }
}

// ---------------- host-side launch ----------------
extern "C" void kernel_launch(void* const* d_in, const int* in_sizes, int n_in,
                              void* d_out, int out_size, void* d_ws, size_t ws_size,
                              hipStream_t stream) {
  const float* X  = (const float*)d_in[0];   // [B,T,D]
  const float* gk = (const float*)d_in[1];   // [D+H, 2H]
  const float* gb = (const float*)d_in[2];   // [2H]
  const float* ck = (const float*)d_in[3];   // [D+H, H]
  const float* cb = (const float*)d_in[4];   // [H]
  float* out = (float*)d_out;                // [B,T,H]

  char* ws = (char*)d_ws;
  __bf16*   WgT   = (__bf16*)(ws + OFF_WGT);
  __bf16*   WcT   = (__bf16*)(ws + OFF_WCT);
  __bf16*   xswz  = (__bf16*)(ws + OFF_XSWZ);
  __bf16*   hswz  = (__bf16*)(ws + OFF_HSWZ);
  __bf16*   rhswz = (__bf16*)(ws + OFF_RHSWZ);
  float*    hbuf  = (float*)(ws + OFF_HBUF);
  float*    ubuf  = (float*)(ws + OFF_UBUF);
  unsigned* bar   = (unsigned*)(ws + OFF_BAR);

  gru_init<<<(BB * HH + NTHREADS - 1) / NTHREADS, NTHREADS, 0, stream>>>(hbuf, hswz, bar);

  const int wtot = NGATE * KTOT + HH * KTOT;
  gru_prep_weights<<<(wtot + NTHREADS - 1) / NTHREADS, NTHREADS, 0, stream>>>(gk, ck, WgT, WcT);

  gru_swizzle_x<<<(BB * TT * DD) / NTHREADS, NTHREADS, 0, stream>>>(X, xswz);

  gru_persistent<<<GRID_WGS, NTHREADS, 0, stream>>>(
      xswz, WgT, WcT, gb, cb, hbuf, hswz, rhswz, ubuf, bar, out);
}
